// Model_1855425871988
// MI455X (gfx1250) — matrix operand order repaired and checked by reading, not compiled, not run on hardware
//
#include <hip/hip_runtime.h>
#include <cstdint>
#include <cstddef>

typedef __attribute__((ext_vector_type(16))) _Float16 v16h;
typedef __attribute__((ext_vector_type(8)))  float    v8f;

#define GNN_N_NODES   100000
#define GNN_N_EDGES   250000
#define GNN_N_GRAPHS  4000
#define GNN_D_RAW     119
#define GNN_D_RAW_PAD 128
#define GNN_D_HID     300
#define GNN_D_PAD     320
#define GNN_NT_FULL   20     // 320/16 column tiles
#define GNN_NT_GRP    4      // column tiles per wave
#define GNN_NG        5      // column-tile groups (20/4)
#define GNN_EB        (GNN_N_EDGES / 16)   // 15625 edge blocks
#define GNN_MB        (GNN_N_NODES / 16)   // 6250 node blocks

union AFragU { v16h v; uint4 q[2]; };

// ---------------------------------------------------------------- utilities
__global__ void zero_f32_k(float* __restrict__ p, int n4) {
  int i = blockIdx.x * blockDim.x + threadIdx.x;
  if (i < n4) ((float4*)p)[i] = make_float4(0.f, 0.f, 0.f, 0.f);
}

// node_feats f32 [N,119] -> f16 [N,128] zero-padded
__global__ void cvt_feats_k(const float* __restrict__ x, _Float16* __restrict__ out) {
  int i = blockIdx.x * blockDim.x + threadIdx.x;
  if (i >= GNN_N_NODES * GNN_D_RAW_PAD) return;
  int n = i >> 7;
  int k = i & (GNN_D_RAW_PAD - 1);
  out[i] = (_Float16)(k < GNN_D_RAW ? x[n * GNN_D_RAW + k] : 0.f);
}

// Pack W f32 [K,Nsrc] row-major into WMMA B-fragment layout, f16, zero-padded.
// blk=(kt*20+nt); within blk: lane(0..31) x half(0..15) contiguous.
// hf=lane>>4: K = kt*32 + 8*hf + h (h<8), kt*32 + 16 + 8*hf + (h-8) (h>=8); N = nt*16+(lane&15)
__global__ void pack_weight_k(const float* __restrict__ W, _Float16* __restrict__ out,
                              int K, int Nsrc, int total) {
  int i = blockIdx.x * blockDim.x + threadIdx.x;
  if (i >= total) return;
  int h    = i & 15;
  int lane = (i >> 4) & 31;
  int blk  = i >> 9;
  int nt   = blk % GNN_NT_FULL;
  int kt   = blk / GNN_NT_FULL;
  int n = nt * 16 + (lane & 15);
  int k = kt * 32 + (lane >> 4) * 8 + (h & 7) + (h >> 3) * 16;
  out[i] = (_Float16)((k < K && n < Nsrc) ? W[k * Nsrc + n] : 0.f);
}

// ------------------------------------------------- dense node GEMM (lift/out)
// One wave computes a 32(M) x 64(N) block: 2 M-blocks x 4 N-tiles.
// A: f16 [N,KPAD] or f32 [N,KPAD]. Writes H f16 [N,320] (padding cols -> 0).
template <int KPAD, bool A_IS_HALF, bool RELU>
__global__ void gemm_node_k(const void* __restrict__ Aptr, const _Float16* __restrict__ Bp,
                            const float* __restrict__ bias, _Float16* __restrict__ H) {
  const int lane = threadIdx.x;          // 0..31
  const int ng   = blockIdx.x;           // 0..4  (N-tile group of 4)
  const int mbg  = blockIdx.y;           // 0..3124 (pair of node blocks)
  const int m    = lane & 15;
  const int hi   = lane >> 4;
  const int r0   = (mbg * 2) * 16 + m;
  const int r1   = (mbg * 2 + 1) * 16 + m;

  v8f c0[GNN_NT_GRP] = {};
  v8f c1[GNN_NT_GRP] = {};
#pragma unroll
  for (int kt = 0; kt < (KPAD >> 5); ++kt) {
    AFragU a0, a1;
    if (A_IS_HALF) {
      const _Float16* p0 = (const _Float16*)Aptr + (size_t)r0 * KPAD + kt * 32 + hi * 8;
      const _Float16* p1 = (const _Float16*)Aptr + (size_t)r1 * KPAD + kt * 32 + hi * 8;
      a0.q[0] = *(const uint4*)p0;  a0.q[1] = *(const uint4*)(p0 + 16);
      a1.q[0] = *(const uint4*)p1;  a1.q[1] = *(const uint4*)(p1 + 16);
    } else {
      const float* p0 = (const float*)Aptr + (size_t)r0 * KPAD + kt * 32 + hi * 8;
      const float* p1 = (const float*)Aptr + (size_t)r1 * KPAD + kt * 32 + hi * 8;
#pragma unroll
      for (int t = 0; t < 8; ++t) {
        a0.v[t] = (_Float16)p0[t];  a0.v[t + 8] = (_Float16)p0[t + 16];
        a1.v[t] = (_Float16)p1[t];  a1.v[t + 8] = (_Float16)p1[t + 16];
      }
    }
#pragma unroll
    for (int t = 0; t < GNN_NT_GRP; ++t) {
      const v16h b = *(const v16h*)(Bp + ((size_t)(kt * GNN_NT_FULL + ng * GNN_NT_GRP + t) * 32 + lane) * 16);
      c0[t] = __builtin_amdgcn_wmma_f32_16x16x32_f16(false, a0.v, false, b, (short)0, c0[t], false, false);
      c1[t] = __builtin_amdgcn_wmma_f32_16x16x32_f16(false, a1.v, false, b, (short)0, c1[t], false, false);
    }
  }

#pragma unroll
  for (int t = 0; t < GNN_NT_GRP; ++t) {
    const int col = (ng * GNN_NT_GRP + t) * 16 + m;
    const float bv = (col < GNN_D_HID) ? bias[col] : 0.f;
    _Float16* __restrict__ o0 = H + (size_t)(mbg * 32) * GNN_D_PAD + col;
    _Float16* __restrict__ o1 = H + (size_t)(mbg * 32 + 16) * GNN_D_PAD + col;
#pragma unroll
    for (int j = 0; j < 8; ++j) {
      int M = j + hi * 8;               // C layout: lanes 0-15 -> M=j, lanes 16-31 -> M=j+8
      float v0 = c0[t][j] + bv;
      float v1 = c1[t][j] + bv;
      if (RELU) { v0 = fmaxf(v0, 0.f); v1 = fmaxf(v1, 0.f); }
      o0[(size_t)M * GNN_D_PAD] = (_Float16)v0;
      o1[(size_t)M * GNN_D_PAD] = (_Float16)v1;
    }
  }
}

// ------------------------------------- fused gather -> GEMM -> ReLU -> scatter
// One wave: 2 edge blocks (32 edges) x 4 N-tiles. 15625 blocks is odd ->
// last wave clamps its 2nd block and suppresses its atomics (uniform branch).
__global__ void msg_scatter_k(const _Float16* __restrict__ H, const _Float16* __restrict__ Bp,
                              const float* __restrict__ bias, const int* __restrict__ src,
                              const int* __restrict__ dst, float* __restrict__ agg) {
  const int lane = threadIdx.x;          // 0..31
  const int ng   = blockIdx.x;           // 0..4
  const int ebg  = blockIdx.y;           // 0..7812
  const int m    = lane & 15;
  const int hi   = lane >> 4;

  const int eb0  = ebg * 2;
  const int eb1r = eb0 + 1;
  const int eb1  = (eb1r < GNN_EB) ? eb1r : (GNN_EB - 1);

  const int s0 = src[eb0 * 16 + m];
  const int s1 = src[eb1 * 16 + m];
  const _Float16* a0row = H + (size_t)s0 * GNN_D_PAD + hi * 8;
  const _Float16* a1row = H + (size_t)s1 * GNN_D_PAD + hi * 8;

  v8f c0[GNN_NT_GRP] = {};
  v8f c1[GNN_NT_GRP] = {};
#pragma unroll
  for (int kt = 0; kt < (GNN_D_PAD >> 5); ++kt) {
    AFragU a0, a1;
    a0.q[0] = *(const uint4*)(a0row + kt * 32);  a0.q[1] = *(const uint4*)(a0row + kt * 32 + 16);
    a1.q[0] = *(const uint4*)(a1row + kt * 32);  a1.q[1] = *(const uint4*)(a1row + kt * 32 + 16);
#pragma unroll
    for (int t = 0; t < GNN_NT_GRP; ++t) {
      const v16h b = *(const v16h*)(Bp + ((size_t)(kt * GNN_NT_FULL + ng * GNN_NT_GRP + t) * 32 + lane) * 16);
      c0[t] = __builtin_amdgcn_wmma_f32_16x16x32_f16(false, a0.v, false, b, (short)0, c0[t], false, false);
      c1[t] = __builtin_amdgcn_wmma_f32_16x16x32_f16(false, a1.v, false, b, (short)0, c1[t], false, false);
    }
  }

  int d0[8], d1[8];
#pragma unroll
  for (int j = 0; j < 8; ++j) {
    d0[j] = dst[eb0 * 16 + hi * 8 + j];
    d1[j] = dst[eb1 * 16 + hi * 8 + j];
  }

#pragma unroll
  for (int t = 0; t < GNN_NT_GRP; ++t) {
    const int col = (ng * GNN_NT_GRP + t) * 16 + m;
    if (col >= GNN_D_HID + 4) continue;           // N-tile 19 is pure padding (uniform skip)
    const float bv = (col < GNN_D_HID) ? bias[col] : 0.f;
#pragma unroll
    for (int j = 0; j < 8; ++j)
      atomicAdd(agg + (size_t)d0[j] * GNN_D_PAD + col, fmaxf(c0[t][j] + bv, 0.f));
    if (eb1r < GNN_EB) {
#pragma unroll
      for (int j = 0; j < 8; ++j)
        atomicAdd(agg + (size_t)d1[j] * GNN_D_PAD + col, fmaxf(c1[t][j] + bv, 0.f));
    }
  }
}

// ------------------------------------------------------------------ readout
__global__ void readout_k(const _Float16* __restrict__ H, const float* __restrict__ Wr,
                          const float* __restrict__ br, const int* __restrict__ gid,
                          float* __restrict__ out) {
  int node = blockIdx.x * blockDim.x + threadIdx.x;
  if (node >= GNN_N_NODES) return;
  const _Float16* row = H + (size_t)node * GNN_D_PAD;
  float a0 = br[0], a1 = br[1];
  // 37 * 8 = 296 vectorized, then 4-element scalar tail (W_read has only 300 rows)
  for (int kb = 0; kb < 296; kb += 8) {
    union { uint4 q; _Float16 h[8]; } u;
    u.q = *(const uint4*)(row + kb);
#pragma unroll
    for (int t = 0; t < 8; ++t) {
      float hv = (float)u.h[t];
      a0 += hv * Wr[2 * (kb + t)];
      a1 += hv * Wr[2 * (kb + t) + 1];
    }
  }
#pragma unroll
  for (int k = 296; k < GNN_D_HID; ++k) {
    float hv = (float)row[k];
    a0 += hv * Wr[2 * k];
    a1 += hv * Wr[2 * k + 1];
  }
  int g = gid[node];
  atomicAdd(out + 2 * g, a0);
  atomicAdd(out + 2 * g + 1, a1);
}

// ------------------------------------------------------------------ launcher
extern "C" void kernel_launch(void* const* d_in, const int* in_sizes, int n_in,
                              void* d_out, int out_size, void* d_ws, size_t ws_size,
                              hipStream_t stream) {
  const float* node_feats = (const float*)d_in[0];
  const float* W_lift     = (const float*)d_in[1];
  const float* b_lift     = (const float*)d_in[2];
  const float* W_msg      = (const float*)d_in[3];
  const float* b_msg      = (const float*)d_in[4];
  const float* W_out      = (const float*)d_in[5];
  const float* b_out      = (const float*)d_in[6];
  const float* W_read     = (const float*)d_in[7];
  const float* b_read     = (const float*)d_in[8];
  const int*   src        = (const int*)d_in[9];
  const int*   dst        = (const int*)d_in[10];
  const int*   gid        = (const int*)d_in[11];
  float* out = (float*)d_out;

  // workspace layout (all offsets 32B-aligned)
  char* ws = (char*)d_ws;
  size_t off = 0;
  _Float16* x16 = (_Float16*)(ws + off); off += (size_t)GNN_N_NODES * GNN_D_RAW_PAD * 2;  // 25.6 MB
  _Float16* h16 = (_Float16*)(ws + off); off += (size_t)GNN_N_NODES * GNN_D_PAD * 2;      // 64 MB
  float*    agg = (float*)(ws + off);    off += (size_t)GNN_N_NODES * GNN_D_PAD * 4;      // 128 MB
  _Float16* wlift_p = (_Float16*)(ws + off); off += (size_t)GNN_D_RAW_PAD * GNN_D_PAD * 2;
  _Float16* wmsg_p  = (_Float16*)(ws + off); off += (size_t)3 * GNN_D_PAD * GNN_D_PAD * 2;
  _Float16* wout_p  = (_Float16*)(ws + off); off += (size_t)3 * GNN_D_PAD * GNN_D_PAD * 2;

  // --- prep: feature conversion + weight packing into WMMA B layout
  {
    int tot = GNN_N_NODES * GNN_D_RAW_PAD;
    cvt_feats_k<<<(tot + 255) / 256, 256, 0, stream>>>(node_feats, x16);
  }
  {
    int tot = GNN_D_RAW_PAD * GNN_D_PAD;
    pack_weight_k<<<(tot + 255) / 256, 256, 0, stream>>>(W_lift, wlift_p, GNN_D_RAW, GNN_D_HID, tot);
  }
  for (int i = 0; i < 3; ++i) {
    int tot = GNN_D_PAD * GNN_D_PAD;
    pack_weight_k<<<(tot + 255) / 256, 256, 0, stream>>>(
        W_msg + (size_t)i * GNN_D_HID * GNN_D_HID, wmsg_p + (size_t)i * GNN_D_PAD * GNN_D_PAD,
        GNN_D_HID, GNN_D_HID, tot);
    pack_weight_k<<<(tot + 255) / 256, 256, 0, stream>>>(
        W_out + (size_t)i * GNN_D_HID * GNN_D_HID, wout_p + (size_t)i * GNN_D_PAD * GNN_D_PAD,
        GNN_D_HID, GNN_D_HID, tot);
  }

  // --- lift: h = x @ W_lift + b (no activation)
  gemm_node_k<GNN_D_RAW_PAD, true, false>
      <<<dim3(GNN_NG, GNN_MB / 2), 32, 0, stream>>>(x16, wlift_p, b_lift, h16);

  // --- 3 rounds of message passing
  for (int i = 0; i < 3; ++i) {
    int n4 = GNN_N_NODES * GNN_D_PAD / 4;
    zero_f32_k<<<(n4 + 255) / 256, 256, 0, stream>>>(agg, n4);

    msg_scatter_k<<<dim3(GNN_NG, (GNN_EB + 1) / 2), 32, 0, stream>>>(
        h16, wmsg_p + (size_t)i * GNN_D_PAD * GNN_D_PAD, b_msg + i * GNN_D_HID, src, dst, agg);

    gemm_node_k<GNN_D_PAD, false, true>
        <<<dim3(GNN_NG, GNN_MB / 2), 32, 0, stream>>>(
            agg, wout_p + (size_t)i * GNN_D_PAD * GNN_D_PAD, b_out + i * GNN_D_HID, h16);
  }

  // --- readout + graph segment-sum
  {
    int n4 = 2 * GNN_N_GRAPHS / 4;
    zero_f32_k<<<(n4 + 255) / 256, 256, 0, stream>>>(out, n4);
  }
  readout_k<<<(GNN_N_NODES + 255) / 256, 256, 0, stream>>>(h16, W_read, b_read, gid, out);
}
